// HSG_X_HWFEBlock_55104430407778
// MI455X (gfx1250) — matrix operand order repaired and checked by reading, not compiled, not run on hardware
//
#include <hip/hip_runtime.h>

// ---------------------------------------------------------------------------
// CDNA5 (gfx1250) implementation. wave32. Matrix ops via
// __builtin_amdgcn_wmma_f32_16x16x32_bf16 (D = A(16x32) * B(32x16) + C).
// ---------------------------------------------------------------------------

typedef __attribute__((ext_vector_type(8)))  float  v8f;
typedef __attribute__((ext_vector_type(16))) __bf16 v16bf;

union Frag32 {                    // one 16x32 bf16 wmma operand per lane
  v16bf          v;
  unsigned short u[16];
  unsigned int   d[8];
  uint4          q[2];
};

// hardware bf16 convert (fptrunc f32->bf16 -> v_cvt_pk_bf16_f32), packed pair
__device__ __forceinline__ unsigned int pack2bf(float a, float b) {
  union { __bf16 h[2]; unsigned int u; } c;
  c.h[0] = (__bf16)a;
  c.h[1] = (__bf16)b;
  return c.u;
}
// (lo16(lo),lo16(hi)) and (hi16(lo),hi16(hi)) via single v_perm_b32 each
__device__ __forceinline__ void interleave16(unsigned int lo, unsigned int hi,
                                             unsigned int& d0, unsigned int& d1) {
  d0 = __builtin_amdgcn_perm(hi, lo, 0x05040100u);
  d1 = __builtin_amdgcn_perm(hi, lo, 0x07060302u);
}
__device__ __forceinline__ float sigm(float x) { return 1.0f / (1.0f + __expf(-x)); }

__device__ __forceinline__ v8f wmma_bf16(const Frag32& a, const Frag32& b, v8f c) {
  return __builtin_amdgcn_wmma_f32_16x16x32_bf16(false, a.v, false, b.v,
                                                 (short)0, c, false, false);
}

// ---- gfx1250 async global->LDS copy (ASYNCcnt path), with sync fallback ----
#if __has_builtin(__builtin_amdgcn_global_load_async_to_lds_b128) && \
    __has_builtin(__builtin_amdgcn_s_wait_asynccnt)
#define USE_ASYNC_LDS 1
// builtin signature (from probe): void(int4 AS1* src_global, int4 AS3* dst_lds,
//                                      imm offset, imm cpol)
typedef int i32x4 __attribute__((vector_size(16)));
typedef __attribute__((address_space(1))) i32x4 i32x4_g;
typedef __attribute__((address_space(3))) i32x4 i32x4_l;
__device__ __forceinline__ void async_copy16(const unsigned short* g,
                                             unsigned int* l) {
  __builtin_amdgcn_global_load_async_to_lds_b128((i32x4_g*)g, (i32x4_l*)l,
                                                 0, 0);
}
__device__ __forceinline__ void async_wait0() {
  __builtin_amdgcn_s_wait_asynccnt(0);
}
#else
#define USE_ASYNC_LDS 0
__device__ __forceinline__ void async_copy16(const unsigned short* g,
                                             unsigned int* l) {
  *(uint4*)l = *(const uint4*)g;
}
__device__ __forceinline__ void async_wait0() {}
#endif

// ---------------------------------------------------------------------------
// K0: f32 -> bf16 packed conversion (fin_w)
// ---------------------------------------------------------------------------
__global__ void k_cvt_bf16(const float* __restrict__ src,
                           unsigned int* __restrict__ dst, int npairs) {
  int i = blockIdx.x * blockDim.x + threadIdx.x;
  const int stride = gridDim.x * blockDim.x;
  for (; i < npairs; i += stride) dst[i] = pack2bf(src[2 * i], src[2 * i + 1]);
}

// ---------------------------------------------------------------------------
// K1: red conv1x1 (256 -> 16) + BN + ReLU as WMMA GEMM.
//   A(16x32) = gate activations (M = hw positions), B(32x16) = red_w^T chunk.
//   Output r_t[(b*4096 + hw)*16 + c]  (hw-major, 16 channels contiguous).
// Block: 128 threads / 4 waves, tile M=64, N=16, K-chunks of 32, dbl-buffered.
// ---------------------------------------------------------------------------
__global__ __launch_bounds__(128) void k_red_gemm(
    const float* __restrict__ x, const float* __restrict__ red_w,
    const float* __restrict__ red_bias,
    const float* __restrict__ bn_g, const float* __restrict__ bn_b,
    const float* __restrict__ bn_m, const float* __restrict__ bn_v,
    float* __restrict__ r_t)
{
  __shared__ unsigned int lds_a[2][64][20];   // [m][kpair], 16 data + 4 pad dw
  __shared__ unsigned int lds_bp[2][16][16];  // [k/2][n] packed (k,k+1) bf16
  const int blk = blockIdx.x;
  const int b   = blk >> 6;                   // 64 blocks per batch
  const int hw0 = (blk & 63) << 6;
  const int t    = threadIdx.x;
  const int wave = t >> 5, lane = t & 31;
  const float* gate = x + ((size_t)b * 512 + 256) * 4096;

  v8f acc;
  #pragma unroll
  for (int i = 0; i < 8; ++i) acc[i] = 0.0f;

  auto stage = [&](int kc, int buf) {
    const int c0 = kc * 32;
    // A: 64 hw x 32 c; thread loads rows (c0+2kp, c0+2kp+1) x 8 hw.
    // Convert same-vector pairs (clean cvt_pk), interleave halves with perm.
    const int kp = t >> 3;                    // 0..15
    const int m0 = (t & 7) * 8;
    const float* pe = gate + (size_t)(c0 + 2 * kp) * 4096 + hw0 + m0;
    const float* po = pe + 4096;
    float4 e0 = *(const float4*)(pe);
    float4 e1 = *(const float4*)(pe + 4);
    float4 o0 = *(const float4*)(po);
    float4 o1 = *(const float4*)(po + 4);
    unsigned int E01 = pack2bf(e0.x, e0.y), O01 = pack2bf(o0.x, o0.y);
    unsigned int E23 = pack2bf(e0.z, e0.w), O23 = pack2bf(o0.z, o0.w);
    unsigned int E45 = pack2bf(e1.x, e1.y), O45 = pack2bf(o1.x, o1.y);
    unsigned int E67 = pack2bf(e1.z, e1.w), O67 = pack2bf(o1.z, o1.w);
    unsigned int d0, d1;
    interleave16(E01, O01, d0, d1);
    lds_a[buf][m0 + 0][kp] = d0; lds_a[buf][m0 + 1][kp] = d1;
    interleave16(E23, O23, d0, d1);
    lds_a[buf][m0 + 2][kp] = d0; lds_a[buf][m0 + 3][kp] = d1;
    interleave16(E45, O45, d0, d1);
    lds_a[buf][m0 + 4][kp] = d0; lds_a[buf][m0 + 5][kp] = d1;
    interleave16(E67, O67, d0, d1);
    lds_a[buf][m0 + 6][kp] = d0; lds_a[buf][m0 + 7][kp] = d1;
    // B: red_w (16,256) transposed chunk, packed k-pairs
    for (int i = t; i < 256; i += 128) {
      const int kk = i >> 4, n = i & 15;
      const int c  = c0 + kk * 2;
      lds_bp[buf][kk][n] = pack2bf(red_w[n * 256 + c], red_w[n * 256 + c + 1]);
    }
  };

  stage(0, 0);
  for (int kc = 0; kc < 8; ++kc) {
    __syncthreads();
    const int buf = kc & 1;
    Frag32 af, bfr;
    const int row = wave * 16 + (lane & 15);
    const int cb  = (lane >> 4) * 4;    // lanes 0-15: K{0-7,16-23}; 16-31: {8-15,24-31}
    af.q[0] = *(const uint4*)&lds_a[buf][row][cb];
    af.q[1] = *(const uint4*)&lds_a[buf][row][cb + 8];
    const int kko = (lane >> 4) * 4;
    const int n   = lane & 15;
    #pragma unroll
    for (int g = 0; g < 8; ++g) bfr.d[g] = lds_bp[buf][kko + g + (g & 4)][n];
    if (kc < 7) stage(kc + 1, buf ^ 1); // overlap next-chunk loads with WMMA
    acc = wmma_bf16(af, bfr, acc);
  }
  // fused BN + ReLU epilogue; D layout: vgpr j -> M = j (+8 for upper half)
  const int n  = lane & 15;
  const float sc = bn_g[n] * rsqrtf(bn_v[n] + 1e-5f);
  const float sh = bn_b[n] + (red_bias[n] - bn_m[n]) * sc;
  const int mb = wave * 16 + ((lane >> 4) * 8);
  #pragma unroll
  for (int j = 0; j < 8; ++j) {
    float v = acc[j] * sc + sh;
    v = v > 0.0f ? v : 0.0f;
    r_t[((size_t)b * 4096 + hw0 + mb + j) * 16 + n] = v;
  }
}

// ---------------------------------------------------------------------------
// K2: per-batch statistics. One block per batch.
//   ctx (=mean r), GCT MLP -> wgct[16], attn = softmax(ctx@fc^T + b)
//   (HWFE branch collapses algebraically: mean(X_re) == ctx),
//   row/col means -> hf/vf convs -> relA[b][c][w], relV[b][c][h].
// ---------------------------------------------------------------------------
__global__ __launch_bounds__(256) void k_stats(
    const float* __restrict__ r_t,
    const float* __restrict__ fc1_w, const float* __restrict__ fc1_b,
    const float* __restrict__ ln_g,  const float* __restrict__ ln_b,
    const float* __restrict__ fc2_w, const float* __restrict__ fc2_b,
    const float* __restrict__ rhw,  const float* __restrict__ rhb,
    const float* __restrict__ rhg,  const float* __restrict__ rhbb,
    const float* __restrict__ rhm,  const float* __restrict__ rhv,
    const float* __restrict__ rvw,  const float* __restrict__ rvb,
    const float* __restrict__ rvg,  const float* __restrict__ rvbb,
    const float* __restrict__ rvm,  const float* __restrict__ rvv,
    const float* __restrict__ fus_w,
    const float* __restrict__ fc_w, const float* __restrict__ fc_b,
    float* __restrict__ wgct, float* __restrict__ attn,
    float* __restrict__ relA, float* __restrict__ relV)
{
  __shared__ float rowm[64][16];   // mean over W  (function of h)
  __shared__ float colm[64][16];   // mean over H  (function of w)
  __shared__ float ctx[16];
  __shared__ float hf[8][64];
  __shared__ float vf[8][64];
  __shared__ float h1n[4];
  __shared__ float gbuf[16];
  const int b = blockIdx.x;
  const int t = threadIdx.x;
  const float* rb = r_t + (size_t)b * 4096 * 16;

  for (int i = t; i < 1024; i += 256) {
    const int h = i >> 4, c = i & 15;
    float s = 0.0f;
    for (int w = 0; w < 64; ++w) s += rb[(h * 64 + w) * 16 + c];
    rowm[h][c] = s * (1.0f / 64.0f);
  }
  for (int i = t; i < 1024; i += 256) {
    const int w = i >> 4, c = i & 15;
    float s = 0.0f;
    for (int h = 0; h < 64; ++h) s += rb[(h * 64 + w) * 16 + c];
    colm[w][c] = s * (1.0f / 64.0f);
  }
  __syncthreads();
  if (t < 16) {
    float s = 0.0f;
    for (int h = 0; h < 64; ++h) s += rowm[h][t];
    ctx[t] = s * (1.0f / 64.0f);
  }
  __syncthreads();
  // --- GCT MLP ---
  if (t < 4) {
    float s = fc1_b[t];
    for (int c = 0; c < 16; ++c) s += ctx[c] * fc1_w[t * 16 + c];
    h1n[t] = s;
  }
  __syncthreads();
  if (t == 0) {
    float mu = 0.25f * (h1n[0] + h1n[1] + h1n[2] + h1n[3]);
    float var = 0.0f;
    for (int j = 0; j < 4; ++j) { float d = h1n[j] - mu; var += d * d; }
    float inv = rsqrtf(var * 0.25f + 1e-5f);
    for (int j = 0; j < 4; ++j) {
      float v = (h1n[j] - mu) * inv * ln_g[j] + ln_b[j];
      h1n[j] = v > 0.0f ? v : 0.0f;
    }
  }
  __syncthreads();
  if (t < 16) {
    float s = fc2_b[t];
    for (int j = 0; j < 4; ++j) s += h1n[j] * fc2_w[t * 4 + j];
    wgct[b * 16 + t] = sigm(s);
    float g = fc_b[t];                       // attn logits from ctx (HWFE collapse)
    for (int c = 0; c < 16; ++c) g += ctx[c] * fc_w[t * 16 + c];
    gbuf[t] = g;
  }
  __syncthreads();
  if (t == 0) {
    float mx = gbuf[0];
    for (int o = 1; o < 16; ++o) mx = fmaxf(mx, gbuf[o]);
    float se = 0.0f;
    for (int o = 0; o < 16; ++o) se += __expf(gbuf[o] - mx);
    const float inv = 1.0f / se;
    for (int o = 0; o < 16; ++o) attn[b * 16 + o] = __expf(gbuf[o] - mx) * inv;
  }
  // --- rel: 1x3 conv over w (hf) and 3x1 conv over h (vf), BN + ReLU ---
  for (int i = t; i < 512; i += 256) {
    const int ch = i >> 6, w = i & 63;
    float s = rhb[ch];
    for (int dx = -1; dx <= 1; ++dx) {
      const int wq = w + dx;
      if (wq >= 0 && wq < 64)
        for (int c = 0; c < 16; ++c)
          s += colm[wq][c] * rhw[ch * 48 + c * 3 + (dx + 1)];
    }
    const float sc = rhg[ch] * rsqrtf(rhv[ch] + 1e-5f);
    float v = (s - rhm[ch]) * sc + rhbb[ch];
    hf[ch][w] = v > 0.0f ? v : 0.0f;
  }
  for (int i = t; i < 512; i += 256) {
    const int ch = i >> 6, h = i & 63;
    float s = rvb[ch];
    for (int dy = -1; dy <= 1; ++dy) {
      const int hq = h + dy;
      if (hq >= 0 && hq < 64)
        for (int c = 0; c < 16; ++c)
          s += rowm[hq][c] * rvw[ch * 48 + c * 3 + (dy + 1)];
    }
    const float sc = rvg[ch] * rsqrtf(rvv[ch] + 1e-5f);
    float v = (s - rvm[ch]) * sc + rvbb[ch];
    vf[ch][h] = v > 0.0f ? v : 0.0f;
  }
  __syncthreads();
  // rel_fus split into w-dependent (hf part) + h-dependent (vf part) tables
  for (int i = t; i < 1024; i += 256) {
    const int o = i >> 6, q = i & 63;
    float sa = 0.0f, sv = 0.0f;
    for (int ch = 0; ch < 8; ++ch) {
      sa += hf[ch][q] * fus_w[o * 16 + ch];
      sv += vf[ch][q] * fus_w[o * 16 + 8 + ch];
    }
    relA[(b * 16 + o) * 64 + q] = sa;
    relV[(b * 16 + o) * 64 + q] = sv;
  }
}

// ---------------------------------------------------------------------------
// K3: interaction = r^2*wgct*rel_map + r*attn, sf conv1x1 (16->256) + BN +
//     sigmoid gate, concat with identity, convert all to bf16 GEMM input.
// Block: 64 contiguous hw (one image row) x all 512 channels.
// Each thread handles 4 consecutive pixels: float4 loads, 8B bf16 stores.
// ---------------------------------------------------------------------------
__global__ __launch_bounds__(256) void k_gate_fuse(
    const float* __restrict__ x,
    const float* __restrict__ r_t,
    const float* __restrict__ wgct, const float* __restrict__ attn,
    const float* __restrict__ relA, const float* __restrict__ relV,
    const float* __restrict__ fus_b,
    const float* __restrict__ sf_w, const float* __restrict__ sf_b,
    const float* __restrict__ bg, const float* __restrict__ bb,
    const float* __restrict__ bm, const float* __restrict__ bv,
    unsigned short* __restrict__ out_bf)
{
  __shared__ float inter[64][17];         // pad 17: conflict-free strided reads
  __shared__ float sfw_s[256 * 16];
  __shared__ float sc_s[256], sh_s[256];
  __shared__ float wg_s[16], at_s[16], fb_s[16];
  const int blk  = blockIdx.x;
  const int b    = blk >> 6;
  const int hrow = blk & 63;              // h coordinate (64 hw = one row)
  const int hw0  = hrow << 6;
  const int t    = threadIdx.x;

  for (int i = t; i < 4096; i += 256) sfw_s[i] = sf_w[i];
  {
    const float s = bg[t] * rsqrtf(bv[t] + 1e-5f);
    sc_s[t] = s;
    sh_s[t] = bb[t] + (sf_b[t] - bm[t]) * s;
  }
  if (t < 16) {
    wg_s[t] = wgct[b * 16 + t];
    at_s[t] = attn[b * 16 + t];
    fb_s[t] = fus_b[t];
  }
  __syncthreads();
  for (int i = t; i < 1024; i += 256) {
    const int p = i >> 4, c = i & 15;     // p = w coordinate
    const float rv = r_t[((size_t)b * 4096 + hw0 + p) * 16 + c];
    const float rm = sigm(relA[(b * 16 + c) * 64 + p] +
                          relV[(b * 16 + c) * 64 + hrow] + fb_s[c]);
    inter[p][c] = rv * rv * wg_s[c] * rm + rv * at_s[c];
  }
  __syncthreads();
  const int p0 = (t & 15) * 4;            // 4 consecutive pixels per thread
  const int cg = t >> 4;                  // 16 channels in parallel
  #pragma unroll 4
  for (int pass = 0; pass < 32; ++pass) {
    const int ch = pass * 16 + cg;        // branch below is wave-uniform
    const size_t idx = ((size_t)b * 512 + ch) * 4096 + hw0 + p0;
    float4 v = *(const float4*)(x + idx);
    if (ch >= 256) {
      const int o = ch - 256;
      float a0 = sh_s[o], a1 = a0, a2 = a0, a3 = a0;
      #pragma unroll
      for (int c = 0; c < 16; ++c) {
        const float w = sfw_s[o * 16 + c];
        a0 += w * inter[p0 + 0][c];
        a1 += w * inter[p0 + 1][c];
        a2 += w * inter[p0 + 2][c];
        a3 += w * inter[p0 + 3][c];
      }
      v.x *= sigm(a0); v.y *= sigm(a1); v.z *= sigm(a2); v.w *= sigm(a3);
    }
    uint2 pk;
    pk.x = pack2bf(v.x, v.y);
    pk.y = pack2bf(v.z, v.w);
    *(uint2*)(out_bf + idx) = pk;
  }
}

// ---------------------------------------------------------------------------
// K4: final conv1x1 (512 -> 512) as bf16 WMMA GEMM, D[o][hw] orientation:
//   A = fin_w (o x c, row-major bf16), B = activations (c x hw, channel-major)
//   -> C/D layout stores are 64B-contiguous in hw: direct NCHW writes.
// Block: 256 threads / 8 waves, tile 64(o) x 128(hw), K = 512 in 16 chunks,
// double-buffered LDS; A tile staged via async global->LDS (ASYNCcnt);
// prefetch of the chunk after next.
// ---------------------------------------------------------------------------
__global__ __launch_bounds__(256) void k_final_gemm(
    const unsigned short* __restrict__ aT,     // (B*512, 4096) bf16 activations
    const unsigned short* __restrict__ fin_bf, // (512, 512) bf16
    const float* __restrict__ fin_b,
    float* __restrict__ out)
{
  __shared__ unsigned int lds_a[2][64][20];    // [o][cpair] A tile (+pad)
  __shared__ unsigned int lds_bp[2][16][132];  // [k/2][hw] packed pairs, padded
  const int blk = blockIdx.x;
  const int b   = blk >> 8;                    // 256 blocks per batch
  const int rem = blk & 255;
  const int o0  = (rem >> 5) * 64;
  const int hw0 = (rem & 31) * 128;
  const int t    = threadIdx.x;
  const int wave = t >> 5, lane = t & 31;
  const int m_sub = wave >> 1;                 // o subtile 0..3
  const int nb    = (wave & 1) * 4;            // hw subtiles nb..nb+3

  v8f acc[4];
  #pragma unroll
  for (int i = 0; i < 4; ++i)
    #pragma unroll
    for (int j = 0; j < 8; ++j) acc[i][j] = 0.0f;

  auto stage = [&](int kc, int buf) {
    const int c0 = kc * 32;
    { // A tile: 64 o x 32 c (bf16, contiguous along c) -> async to LDS
      const int o = t >> 2, part = t & 3;
      async_copy16(&fin_bf[(size_t)(o0 + o) * 512 + c0 + part * 8],
                   &lds_a[buf][o][part * 4]);
    }
    { // B tile: 32 c x 128 hw, interleave (c, c+1) pairs via v_perm_b32
      const int kk = t >> 4, noff = (t & 15) * 8;
      const unsigned short* r0 =
          aT + ((size_t)b * 512 + c0 + 2 * kk) * 4096 + hw0 + noff;
      const unsigned short* r1 = r0 + 4096;
      __builtin_prefetch(r0 + 32 * 4096, 0, 1);   // chunk after next -> L2
      uint4 lo = *(const uint4*)r0;
      uint4 hi = *(const uint4*)r1;
      unsigned int d0, d1;
      interleave16(lo.x, hi.x, d0, d1);
      lds_bp[buf][kk][noff + 0] = d0; lds_bp[buf][kk][noff + 1] = d1;
      interleave16(lo.y, hi.y, d0, d1);
      lds_bp[buf][kk][noff + 2] = d0; lds_bp[buf][kk][noff + 3] = d1;
      interleave16(lo.z, hi.z, d0, d1);
      lds_bp[buf][kk][noff + 4] = d0; lds_bp[buf][kk][noff + 5] = d1;
      interleave16(lo.w, hi.w, d0, d1);
      lds_bp[buf][kk][noff + 6] = d0; lds_bp[buf][kk][noff + 7] = d1;
    }
  };

  stage(0, 0);
  for (int kc = 0; kc < 16; ++kc) {
    async_wait0();                         // async A-tile copies -> LDS done
    __syncthreads();
    const int buf = kc & 1;
    Frag32 af;
    const int arow = m_sub * 16 + (lane & 15);
    const int cb   = (lane >> 4) * 4;
    af.q[0] = *(const uint4*)&lds_a[buf][arow][cb];
    af.q[1] = *(const uint4*)&lds_a[buf][arow][cb + 8];
    const int kko = (lane >> 4) * 4;
    const int nlo = lane & 15;
    Frag32 b0, b1, b2, b3;
    #pragma unroll
    for (int g = 0; g < 8; ++g) {
      b0.d[g] = lds_bp[buf][kko + g + (g & 4)][(nb + 0) * 16 + nlo];
      b1.d[g] = lds_bp[buf][kko + g + (g & 4)][(nb + 1) * 16 + nlo];
      b2.d[g] = lds_bp[buf][kko + g + (g & 4)][(nb + 2) * 16 + nlo];
      b3.d[g] = lds_bp[buf][kko + g + (g & 4)][(nb + 3) * 16 + nlo];
    }
    if (kc < 15) stage(kc + 1, buf ^ 1);   // overlap global loads with WMMA
    acc[0] = wmma_bf16(af, b0, acc[0]);
    acc[1] = wmma_bf16(af, b1, acc[1]);
    acc[2] = wmma_bf16(af, b2, acc[2]);
    acc[3] = wmma_bf16(af, b3, acc[3]);
  }
  #pragma unroll
  for (int i = 0; i < 4; ++i) {
    const int hw = hw0 + (nb + i) * 16 + (lane & 15);
    #pragma unroll
    for (int j = 0; j < 8; ++j) {
      const int o = o0 + m_sub * 16 + j + ((lane >> 4) * 8);
      out[((size_t)b * 512 + o) * 4096 + hw] = acc[i][j] + fin_b[o];
    }
  }
}

// ---------------------------------------------------------------------------
extern "C" void kernel_launch(void* const* d_in, const int* in_sizes, int n_in,
                              void* d_out, int out_size, void* d_ws, size_t ws_size,
                              hipStream_t stream) {
  (void)in_sizes; (void)n_in; (void)out_size; (void)ws_size;
  const float* x     = (const float*)d_in[0];
  const float* red_w = (const float*)d_in[1];

  char* ws = (char*)d_ws;
  float*          r_t    = (float*)(ws + 0);                 // 4,194,304 B
  unsigned short* fin_bf = (unsigned short*)(ws + 4194304);  //   524,288 B
  float*          wgct   = (float*)(ws + 4718592);           //     1,024 B
  float*          attn   = (float*)(ws + 4719616);           //     1,024 B
  float*          relA   = (float*)(ws + 4720640);           //    65,536 B
  float*          relV   = (float*)(ws + 4786176);           //    65,536 B
  unsigned short* out_bf = (unsigned short*)(ws + 4851712);  // 67,108,864 B

  k_cvt_bf16<<<512, 256, 0, stream>>>((const float*)d_in[37],
                                      (unsigned int*)fin_bf, 512 * 512 / 2);

  k_red_gemm<<<1024, 128, 0, stream>>>(
      x, red_w, (const float*)d_in[2],
      (const float*)d_in[3], (const float*)d_in[4],
      (const float*)d_in[5], (const float*)d_in[6], r_t);

  k_stats<<<16, 256, 0, stream>>>(
      r_t,
      (const float*)d_in[7],  (const float*)d_in[8],
      (const float*)d_in[9],  (const float*)d_in[10],
      (const float*)d_in[11], (const float*)d_in[12],
      (const float*)d_in[13], (const float*)d_in[14],
      (const float*)d_in[15], (const float*)d_in[16],
      (const float*)d_in[17], (const float*)d_in[18],
      (const float*)d_in[19], (const float*)d_in[20],
      (const float*)d_in[21], (const float*)d_in[22],
      (const float*)d_in[23], (const float*)d_in[24],
      (const float*)d_in[25],
      (const float*)d_in[29], (const float*)d_in[30],
      wgct, attn, relA, relV);

  k_gate_fuse<<<1024, 256, 0, stream>>>(
      x, r_t, wgct, attn, relA, relV,
      (const float*)d_in[26],
      (const float*)d_in[31], (const float*)d_in[32],
      (const float*)d_in[33], (const float*)d_in[34],
      (const float*)d_in[35], (const float*)d_in[36],
      out_bf);

  k_final_gemm<<<4096, 256, 0, stream>>>(
      out_bf, fin_bf, (const float*)d_in[38], (float*)d_out);
}
